// GAT_L2_intervention_66271345377531
// MI455X (gfx1250) — matrix operand order repaired and checked by reading, not compiled, not run on hardware
//
#include <hip/hip_runtime.h>

// ---------------------------------------------------------------------------
// 2-layer GAT for MI455X (gfx1250, wave32, WMMA).
//  * GEMMs: operands pre-packed into bf16 WMMA-fragment layout, hot loop is
//    pure global_load_b128 -> v_wmma_f32_16x16x32_bf16 (fully unrolled).
//  * Segment softmax: 3-pass atomics on L2-resident buffers (~173 MB total
//    working set < 192 MB global L2).
// ---------------------------------------------------------------------------

typedef __attribute__((ext_vector_type(16))) __bf16       bf16x16;
typedef __attribute__((ext_vector_type(8)))  float        f32x8;
typedef __attribute__((ext_vector_type(8)))  unsigned int u32x8;

union FragCvt { bf16x16 bf; u32x8 u; };

__device__ __forceinline__ float leaky02(float x) { return x > 0.f ? x : 0.2f * x; }

// Monotone float -> uint key so unsigned atomicMax implements float max.
__device__ __forceinline__ unsigned enc_f32(float f) {
    unsigned b = __float_as_uint(f);
    return b ^ ((b & 0x80000000u) ? 0xFFFFFFFFu : 0x80000000u);
}
__device__ __forceinline__ float dec_f32(unsigned u) {
    unsigned b = (u & 0x80000000u) ? (u ^ 0x80000000u) : ~u;
    return __uint_as_float(b);
}

// ---------------------------------------------------------------------------
// Pack A operand: X[rows,K] f32 -> per-(rowTile,kk,lane) 16-bf16 fragment.
// A-layout (ISA 16-bit A 16x32): lane half h: elems 0..7 -> K=kk*32+h*8+i,
// elems 8..15 -> K=kk*32+16+h*8+(i-8); row = rowTile*16 + (lane&15).
// Fragment index fi = (rowTile*ksteps + kk)*32 + lane.
// ---------------------------------------------------------------------------
__global__ void pack_A_bf16(const float* __restrict__ X, u32x8* __restrict__ PA,
                            int rowTiles, int ksteps)
{
    int fi = blockIdx.x * blockDim.x + threadIdx.x;
    int total = rowTiles * ksteps * 32;
    if (fi >= total) return;
    int lane = fi & 31;
    int kk = (fi >> 5) % ksteps;
    int rowTile = fi / (ksteps * 32);
    int half = lane >> 4, l16 = lane & 15;
    int K = ksteps * 32;
    int row = rowTile * 16 + l16;
    int ka0 = kk * 32 + half * 8;
    const float4* ap = reinterpret_cast<const float4*>(X + (size_t)row * K + ka0);
    const float4* aq = reinterpret_cast<const float4*>(X + (size_t)row * K + ka0 + 16);
    float4 a0 = ap[0], a1 = ap[1], a2 = aq[0], a3 = aq[1];
    FragCvt f;
    f.bf[0]=(__bf16)a0.x;  f.bf[1]=(__bf16)a0.y;  f.bf[2]=(__bf16)a0.z;  f.bf[3]=(__bf16)a0.w;
    f.bf[4]=(__bf16)a1.x;  f.bf[5]=(__bf16)a1.y;  f.bf[6]=(__bf16)a1.z;  f.bf[7]=(__bf16)a1.w;
    f.bf[8]=(__bf16)a2.x;  f.bf[9]=(__bf16)a2.y;  f.bf[10]=(__bf16)a2.z; f.bf[11]=(__bf16)a2.w;
    f.bf[12]=(__bf16)a3.x; f.bf[13]=(__bf16)a3.y; f.bf[14]=(__bf16)a3.z; f.bf[15]=(__bf16)a3.w;
    PA[fi] = f.u;
}

// ---------------------------------------------------------------------------
// Pack B operand: W[K,Ncol] f32 -> per-(colTile,kk,lane) 16-bf16 fragment.
// B-layout (ISA 16-bit B 32x16): elem i -> K = kk*32 + (lane>>4)*8 + (i<8 ? i : i+8),
// col = colTile*16 + (lane&15).  fi = (colTile*ksteps + kk)*32 + lane.
// ---------------------------------------------------------------------------
__global__ void pack_B_bf16(const float* __restrict__ W, u32x8* __restrict__ PB,
                            int colTiles, int ksteps, int Ncol)
{
    int fi = blockIdx.x * blockDim.x + threadIdx.x;
    int total = colTiles * ksteps * 32;
    if (fi >= total) return;
    int lane = fi & 31;
    int kk = (fi >> 5) % ksteps;
    int colTile = fi / (ksteps * 32);
    int half = lane >> 4, l16 = lane & 15;
    int col = colTile * 16 + l16;
    int kb0 = kk * 32 + half * 8;
    const float* wp = W + (size_t)kb0 * Ncol + col;
    FragCvt f;
    #pragma unroll
    for (int i = 0; i < 16; ++i) f.bf[i] = (__bf16)wp[(size_t)(i < 8 ? i : i + 8) * Ncol];
    PB[fi] = f.u;
}

// ---------------------------------------------------------------------------
// C[rows, Ncol] = A * B from packed fragments. One 16x16 tile per wave;
// fully unrolled K loop: 2x global_load_b128 per operand per v_wmma.
// ---------------------------------------------------------------------------
template<int KSTEPS>
__global__ void gemm_packed_wmma(const u32x8* __restrict__ PA,
                                 const u32x8* __restrict__ PB,
                                 float* __restrict__ C, int Ncol)
{
    const int lane = threadIdx.x & 31;
    const int wave = threadIdx.x >> 5;
    const int half = lane >> 4, l16 = lane & 15;
    const int rowTile = blockIdx.x;
    const int colTile = blockIdx.y * (blockDim.x >> 5) + wave;

    const u32x8* pa = PA + (size_t)rowTile * KSTEPS * 32 + lane;
    const u32x8* pb = PB + (size_t)colTile * KSTEPS * 32 + lane;

    f32x8 acc = {0.f, 0.f, 0.f, 0.f, 0.f, 0.f, 0.f, 0.f};
    #pragma unroll
    for (int kk = 0; kk < KSTEPS; ++kk) {
        FragCvt a, b;
        a.u = pa[kk * 32];
        b.u = pb[kk * 32];
        acc = __builtin_amdgcn_wmma_f32_16x16x32_bf16(
                  false, a.bf, false, b.bf, (short)0, acc, false, false);
    }

    const int col = colTile * 16 + l16;
    #pragma unroll
    for (int j = 0; j < 8; ++j)
        C[(size_t)(rowTile * 16 + j + 8 * half) * Ncol + col] = acc[j];
}

// ---------------------------------------------------------------------------
// Per-(node, head) attention scores: a_src = sum_c xs*att_src, a_dst likewise.
// ---------------------------------------------------------------------------
__global__ void attn_scores(const float* __restrict__ xs,
                            const float* __restrict__ att_s,
                            const float* __restrict__ att_d,
                            float* __restrict__ as_out,
                            float* __restrict__ ad_out,
                            int n, int H, int Cch)
{
    int t = blockIdx.x * blockDim.x + threadIdx.x;
    if (t >= n * H) return;
    int node = t / H, h = t - node * H;
    const float* xp = xs + ((size_t)node * H + h) * Cch;
    const float* ap = att_s + (size_t)h * Cch;
    const float* bp = att_d + (size_t)h * Cch;
    float s0 = 0.f, s1 = 0.f;
    for (int c = 0; c < Cch; c += 4) {
        float4 xv = *reinterpret_cast<const float4*>(xp + c);
        float4 av = *reinterpret_cast<const float4*>(ap + c);
        float4 bv = *reinterpret_cast<const float4*>(bp + c);
        s0 += xv.x*av.x + xv.y*av.y + xv.z*av.z + xv.w*av.w;
        s1 += xv.x*bv.x + xv.y*bv.y + xv.z*bv.z + xv.w*bv.w;
    }
    as_out[t] = s0;
    ad_out[t] = s1;
}

__global__ void fill_f32(float* __restrict__ p, float v, int count)
{
    int i = blockIdx.x * blockDim.x + threadIdx.x;
    if (i < count) p[i] = v;
}

// ---------------------------------------------------------------------------
// Segment softmax pass 1: per-destination running max (uint-key atomicMax).
// Edges e in [0,E) come from src/dst; e in [E, E+n) are self-loops.
// ---------------------------------------------------------------------------
__global__ void edge_max_kernel(const int* __restrict__ src, const int* __restrict__ dst,
                                const float* __restrict__ as_in, const float* __restrict__ ad_in,
                                unsigned* __restrict__ amaxKey, int E, int n, int H)
{
    int t = blockIdx.x * blockDim.x + threadIdx.x;
    int total = (E + n) * H;
    if (t >= total) return;
    int e = t / H, h = t - e * H;
    int s, d;
    if (e < E) { s = src[e]; d = dst[e]; } else { s = e - E; d = s; }
    float alpha = leaky02(as_in[s * H + h] + ad_in[d * H + h]);
    atomicMax(&amaxKey[d * H + h], enc_f32(alpha));
}

// Pass 2: denom[d,h] += exp(alpha - amax[d,h])
__global__ void edge_denom_kernel(const int* __restrict__ src, const int* __restrict__ dst,
                                  const float* __restrict__ as_in, const float* __restrict__ ad_in,
                                  const unsigned* __restrict__ amaxKey,
                                  float* __restrict__ denom, int E, int n, int H)
{
    int t = blockIdx.x * blockDim.x + threadIdx.x;
    int total = (E + n) * H;
    if (t >= total) return;
    int e = t / H, h = t - e * H;
    int s, d;
    if (e < E) { s = src[e]; d = dst[e]; } else { s = e - E; d = s; }
    int di = d * H + h;
    float alpha = leaky02(as_in[s * H + h] + ad_in[di]);
    atomicAdd(&denom[di], __expf(alpha - dec_f32(amaxKey[di])));
}

// ---------------------------------------------------------------------------
// Pass 3: weighted scatter-add. One wave32 group per edge; lane l computes the
// softmax weight for head (l & (H-1)), broadcast with __shfl. Each lane owns a
// contiguous F/32-element slice (head constant per lane -> coalesced b128
// gathers), then f32 atomic adds into the L2-resident aggregation buffer.
// ---------------------------------------------------------------------------
template<int H, int C>
__global__ void edge_agg_kernel(const int* __restrict__ src, const int* __restrict__ dst,
                                const float* __restrict__ as_in, const float* __restrict__ ad_in,
                                const unsigned* __restrict__ amaxKey,
                                const float* __restrict__ denom,
                                const float* __restrict__ xs, float* __restrict__ agg,
                                int E, int n)
{
    constexpr int F   = H * C;
    constexpr int PER = F / 32;           // contiguous elems per lane (8 or 2)
    int lane = threadIdx.x & 31;
    int e = (blockIdx.x * blockDim.x + threadIdx.x) >> 5;
    if (e >= E + n) return;
    int s, d;
    if (e < E) { s = src[e]; d = dst[e]; } else { s = e - E; d = s; }
    int h = lane & (H - 1);
    int di = d * H + h;
    float alpha = leaky02(as_in[s * H + h] + ad_in[di]);
    float wt = __expf(alpha - dec_f32(amaxKey[di])) / denom[di];

    const int base = lane * PER;          // all PER elems share head = base/C
    float w = __shfl(wt, base / C, 32);
    const float* xp = xs + (size_t)s * F + base;
    float* op = agg + (size_t)d * F + base;
    #pragma unroll
    for (int i = 0; i < PER; ++i)
        atomicAdd(op + i, w * xp[i]);
}

// h = elu(agg + bias) in place (layer-1 epilogue)
__global__ void bias_elu_kernel(float* __restrict__ h, const float* __restrict__ bias,
                                int total, int F)
{
    int i = blockIdx.x * blockDim.x + threadIdx.x;
    if (i >= total) return;
    float v = h[i] + bias[i % F];
    h[i] = v > 0.f ? v : (__expf(v) - 1.f);
}

// out = agg + bias (layer-2 epilogue)
__global__ void out_bias_kernel(const float* __restrict__ agg, const float* __restrict__ bias,
                                float* __restrict__ out, int total, int F)
{
    int i = blockIdx.x * blockDim.x + threadIdx.x;
    if (i >= total) return;
    out[i] = agg[i] + bias[i % F];
}

// ---------------------------------------------------------------------------
extern "C" void kernel_launch(void* const* d_in, const int* in_sizes, int n_in,
                              void* d_out, int out_size, void* d_ws, size_t ws_size,
                              hipStream_t stream)
{
    (void)in_sizes; (void)n_in; (void)out_size; (void)ws_size;

    constexpr int N  = 50000, E = 800000;
    constexpr int H1 = 8,  C1 = 32, F1 = 256;   // layer 1: 8 heads x 32
    constexpr int H2 = 1,  C2 = 64, F2 = 64;    // layer 2: 1 head x 64
    constexpr int RT = N / 16;                  // 3125 row tiles (exact)
    constexpr int KS1 = 128 / 32, KS2 = F1 / 32;

    const float* x     = (const float*)d_in[0];
    const int*   ei    = (const int*)d_in[1];   // edge_index flat: src[E] ++ dst[E]
    const float* W1    = (const float*)d_in[2];
    const float* asrc1 = (const float*)d_in[3];
    const float* adst1 = (const float*)d_in[4];
    const float* b1    = (const float*)d_in[5];
    const float* W2    = (const float*)d_in[6];
    const float* asrc2 = (const float*)d_in[7];
    const float* adst2 = (const float*)d_in[8];
    const float* b2    = (const float*)d_in[9];
    const int* src = ei;
    const int* dst = ei + E;
    float* out = (float*)d_out;

    // Carve workspace (256B-aligned regions).
    char* wsp = (char*)d_ws;
    auto carveB = [&](size_t bytes) -> void* {
        void* p = (void*)wsp;
        wsp += ((bytes + 255) & ~(size_t)255);
        return p;
    };
    float*    xs1  = (float*)carveB((size_t)N * F1 * 4);
    float*    as1  = (float*)carveB((size_t)N * H1 * 4);
    float*    ad1  = (float*)carveB((size_t)N * H1 * 4);
    unsigned* mk1  = (unsigned*)carveB((size_t)N * H1 * 4);
    float*    dn1  = (float*)carveB((size_t)N * H1 * 4);
    float*    agg1 = (float*)carveB((size_t)N * F1 * 4);   // becomes h
    float*    xs2  = (float*)carveB((size_t)N * F2 * 4);
    float*    as2  = (float*)carveB((size_t)N * 4);
    float*    ad2  = (float*)carveB((size_t)N * 4);
    unsigned* mk2  = (unsigned*)carveB((size_t)N * 4);
    float*    dn2  = (float*)carveB((size_t)N * 4);
    float*    agg2 = (float*)carveB((size_t)N * F2 * 4);
    u32x8*    PA1  = (u32x8*)carveB((size_t)RT * KS1 * 32 * 32);
    u32x8*    PB1  = (u32x8*)carveB((size_t)(F1/16) * KS1 * 32 * 32);
    u32x8*    PA2  = (u32x8*)carveB((size_t)RT * KS2 * 32 * 32);
    u32x8*    PB2  = (u32x8*)carveB((size_t)(F2/16) * KS2 * 32 * 32);

    const int TB = 256;
    auto nblk = [](long total, int tb) { return (unsigned)((total + tb - 1) / tb); };

    // ---------------- Layer 1 ----------------
    pack_A_bf16<<<nblk((long)RT * KS1 * 32, TB), TB, 0, stream>>>(x, PA1, RT, KS1);
    pack_B_bf16<<<nblk((long)(F1/16) * KS1 * 32, TB), TB, 0, stream>>>(W1, PB1, F1/16, KS1, F1);
    gemm_packed_wmma<KS1><<<dim3(RT, (F1/16)/4), 128, 0, stream>>>(PA1, PB1, xs1, F1);

    attn_scores<<<nblk((long)N * H1, TB), TB, 0, stream>>>(xs1, asrc1, adst1, as1, ad1, N, H1, C1);
    fill_f32<<<nblk((long)N * H1, TB), TB, 0, stream>>>((float*)mk1, 0.f, N * H1); // key 0 < enc(finite)
    fill_f32<<<nblk((long)N * H1, TB), TB, 0, stream>>>(dn1, 0.f, N * H1);
    fill_f32<<<nblk((long)N * F1, TB), TB, 0, stream>>>(agg1, 0.f, N * F1);
    edge_max_kernel  <<<nblk((long)(E + N) * H1, TB), TB, 0, stream>>>(src, dst, as1, ad1, mk1, E, N, H1);
    edge_denom_kernel<<<nblk((long)(E + N) * H1, TB), TB, 0, stream>>>(src, dst, as1, ad1, mk1, dn1, E, N, H1);
    edge_agg_kernel<H1, C1><<<nblk((long)(E + N) * 32, TB), TB, 0, stream>>>(src, dst, as1, ad1, mk1, dn1,
                                                                             xs1, agg1, E, N);
    bias_elu_kernel<<<nblk((long)N * F1, TB), TB, 0, stream>>>(agg1, b1, N * F1, F1);

    // ---------------- Layer 2 ----------------
    pack_A_bf16<<<nblk((long)RT * KS2 * 32, TB), TB, 0, stream>>>(agg1, PA2, RT, KS2);
    pack_B_bf16<<<nblk((long)(F2/16) * KS2 * 32, TB), TB, 0, stream>>>(W2, PB2, F2/16, KS2, F2);
    gemm_packed_wmma<KS2><<<dim3(RT, 1), 128, 0, stream>>>(PA2, PB2, xs2, F2);

    attn_scores<<<nblk((long)N * H2, TB), TB, 0, stream>>>(xs2, asrc2, adst2, as2, ad2, N, H2, C2);
    fill_f32<<<nblk((long)N, TB), TB, 0, stream>>>((float*)mk2, 0.f, N);
    fill_f32<<<nblk((long)N, TB), TB, 0, stream>>>(dn2, 0.f, N);
    fill_f32<<<nblk((long)N * F2, TB), TB, 0, stream>>>(agg2, 0.f, N * F2);
    edge_max_kernel  <<<nblk((long)(E + N), TB), TB, 0, stream>>>(src, dst, as2, ad2, mk2, E, N, H2);
    edge_denom_kernel<<<nblk((long)(E + N), TB), TB, 0, stream>>>(src, dst, as2, ad2, mk2, dn2, E, N, H2);
    edge_agg_kernel<H2, C2><<<nblk((long)(E + N) * 32, TB), TB, 0, stream>>>(src, dst, as2, ad2, mk2, dn2,
                                                                             xs2, agg2, E, N);
    out_bias_kernel<<<nblk((long)N * F2, TB), TB, 0, stream>>>(agg2, b2, out, N * F2, F2);
}
